// BasicTransformerBlock_77635828842599
// MI455X (gfx1250) — matrix operand order repaired and checked by reading, not compiled, not run on hardware
//
#include <hip/hip_runtime.h>
#include <hip/hip_bf16.h>
#include <math.h>

// ---------------------------------------------------------------------------
// CDNA5 (gfx1250) wave32 WMMA transformer block.
// v_wmma_f32_16x16x32_bf16 everywhere; staging uses clamped-index b128 global
// loads with register double-buffering (next tile fetched during WMMAs),
// packed ds_store_b64 transposes, and b128 fragment loads from LDS.
// ---------------------------------------------------------------------------

typedef __bf16  bf16_t;
typedef __bf16  v16bf __attribute__((ext_vector_type(16)));
typedef __bf16  v4bf  __attribute__((ext_vector_type(4)));
typedef float   v8f   __attribute__((ext_vector_type(8)));

__device__ __forceinline__ v8f v8f_zero() {
    v8f z;
#pragma unroll
    for (int i = 0; i < 8; ++i) z[i] = 0.0f;
    return z;
}

__device__ __forceinline__ v8f wmma_bf16(v16bf a, v16bf b, v8f c) {
    return __builtin_amdgcn_wmma_f32_16x16x32_bf16(
        false, a, false, b, (short)0, c, false, false);
}

// A fragment: 16(M) x 32(K) tile, row-major in LDS -> 2x ds_load_b128.
__device__ __forceinline__ v16bf load_frag_a(const bf16_t* lds, int stride, int lane) {
    const int m     = lane & 15;
    const int khalf = (lane >> 4) << 3;
    v16bf a;
#pragma unroll
    for (int e = 0; e < 16; ++e) {
        const int p = e >> 1;
        const int w = e & 1;
        const int k = ((p & 4) << 2) + khalf + ((p & 3) << 1) + w;
        a[e] = lds[m * stride + k];
    }
    return a;
}

// B fragment from a column-major tile[n][k] -> 2x ds_load_b128.
__device__ __forceinline__ v16bf load_frag_bt(const bf16_t* lds, int stride, int lane) {
    const int n     = lane & 15;
    const int kbase = (lane >> 4) << 3;
    v16bf b;
#pragma unroll
    for (int e = 0; e < 16; ++e) b[e] = lds[n * stride + ((e & 8) << 1) + kbase + (e & 7)];
    return b;
}

__device__ __forceinline__ v4bf cvt4(float4 f) {
    v4bf h;
    h[0] = (bf16_t)f.x; h[1] = (bf16_t)f.y; h[2] = (bf16_t)f.z; h[3] = (bf16_t)f.w;
    return h;
}
__device__ __forceinline__ float f4_at(const float4& v, int j) {
    return (&v.x)[j];
}

// ---------------------------------------------------------------------------
// LayerNorm: one block per row.
// ---------------------------------------------------------------------------
__global__ __launch_bounds__(256)
void layernorm_kernel(const float* __restrict__ X, const float* __restrict__ g,
                      const float* __restrict__ b, float* __restrict__ Y, int C) {
    const int row = blockIdx.x;
    const float* xr = X + (size_t)row * C;
    float*       yr = Y + (size_t)row * C;
    float s = 0.0f, ss = 0.0f;
    for (int c = threadIdx.x; c < C; c += 256) {
        const float v = xr[c];
        s += v; ss += v * v;
    }
    __shared__ float rs[256], rss[256];
    rs[threadIdx.x] = s; rss[threadIdx.x] = ss;
    __syncthreads();
    for (int off = 128; off > 0; off >>= 1) {
        if ((int)threadIdx.x < off) {
            rs[threadIdx.x]  += rs[threadIdx.x + off];
            rss[threadIdx.x] += rss[threadIdx.x + off];
        }
        __syncthreads();
    }
    const float mu  = rs[0] / (float)C;
    const float var = rss[0] / (float)C - mu * mu;
    const float inv = rsqrtf(var + 1e-5f);
    for (int c = threadIdx.x; c < C; c += 256)
        yr[c] = (xr[c] - mu) * inv * g[c] + b[c];
}

// ---------------------------------------------------------------------------
// GEMM: C[M,N] = A[M,K]*B[K,N] (+bias)(+resid). K%32==0, N%128==0.
// 256 thr = 8 waves; block tile 128x128; wave tile 32x64 -> 8 WMMAs/K-step.
// Register double-buffered staging: fetch(k+32) issued before the WMMA block.
// ---------------------------------------------------------------------------
#define BM 128
#define BN 128
#define BK 32
#define AST 40   // 80B row stride: 16B multiple, conflict-free bank walk
#define BST 40

__global__ __launch_bounds__(256)
void gemm_bf16_kernel(const float* __restrict__ A, int lda,
                      const float* __restrict__ B, int ldb,
                      const float* __restrict__ bias,
                      const float* __restrict__ resid,
                      float* __restrict__ C, int ldc,
                      int M, int N, int K) {
    __shared__ __align__(16) bf16_t sA [BM][AST];
    __shared__ __align__(16) bf16_t sBt[BN][BST];

    const int tid  = threadIdx.x;
    const int lane = tid & 31;
    const int wave = tid >> 5;
    const int wr   = wave >> 1;      // 0..3: 32-row strip
    const int wc   = wave & 1;       // 0..1: 64-col strip
    const int m0   = blockIdx.y * BM;
    const int n0   = blockIdx.x * BN;

    const int bq  = tid >> 5;        // B k-row quad 0..7
    const int bc4 = (tid & 31) << 2; // B col 0..124

    v8f acc[2][4];
#pragma unroll
    for (int i = 0; i < 2; ++i)
#pragma unroll
        for (int j = 0; j < 4; ++j) acc[i][j] = v8f_zero();

    float4 ra[4], rb[4];

    auto fetch = [&](int k0) {
#pragma unroll
        for (int it = 0; it < 4; ++it) {
            const int idx = tid + it * 256;
            const int r   = idx >> 3;
            const int c   = (idx & 7) << 2;
            const int rr  = (m0 + r < M) ? (m0 + r) : (M - 1); // clamp, no branch
            ra[it] = *(const float4*)(A + (size_t)rr * lda + k0 + c);
        }
#pragma unroll
        for (int j = 0; j < 4; ++j)
            rb[j] = *(const float4*)(B + (size_t)(k0 + 4 * bq + j) * ldb + n0 + bc4);
    };

    auto stage = [&]() {
#pragma unroll
        for (int it = 0; it < 4; ++it) {
            const int idx = tid + it * 256;
            const int r   = idx >> 3;
            const int c   = (idx & 7) << 2;
            float4 f = ra[it];
            if (m0 + r >= M) f = make_float4(0.f, 0.f, 0.f, 0.f);
            *(v4bf*)&sA[r][c] = cvt4(f);
        }
#pragma unroll
        for (int j = 0; j < 4; ++j) {   // pack 4 k-values per column -> b64
            v4bf h;
            h[0] = (bf16_t)f4_at(rb[0], j);
            h[1] = (bf16_t)f4_at(rb[1], j);
            h[2] = (bf16_t)f4_at(rb[2], j);
            h[3] = (bf16_t)f4_at(rb[3], j);
            *(v4bf*)&sBt[bc4 + j][4 * bq] = h;
        }
    };

    fetch(0);
    for (int k0 = 0; k0 < K; k0 += BK) {
        stage();
        __syncthreads();
        if (k0 + BK < K) fetch(k0 + BK);   // overlap with WMMAs below
        const v16bf a0 = load_frag_a(&sA[wr * 32][0],      AST, lane);
        const v16bf a1 = load_frag_a(&sA[wr * 32 + 16][0], AST, lane);
#pragma unroll
        for (int j = 0; j < 4; ++j) {
            const v16bf b = load_frag_bt(&sBt[wc * 64 + j * 16][0], BST, lane);
            acc[0][j] = wmma_bf16(a0, b, acc[0][j]);
            acc[1][j] = wmma_bf16(a1, b, acc[1][j]);
        }
        __syncthreads();
    }

    const int half = lane >> 4;
    const int ncol = lane & 15;
#pragma unroll
    for (int i = 0; i < 2; ++i)
#pragma unroll
        for (int j = 0; j < 4; ++j)
#pragma unroll
            for (int r = 0; r < 8; ++r) {
                const int gm = m0 + wr * 32 + i * 16 + half * 8 + r;
                const int gn = n0 + wc * 64 + j * 16 + ncol;
                if (gm < M) {
                    float v = acc[i][j][r];
                    if (bias)  v += bias[gn];
                    if (resid) v += resid[(size_t)gm * ldc + gn];
                    C[(size_t)gm * ldc + gn] = v;
                }
            }
}

// ---------------------------------------------------------------------------
// Flash attention: grid = (Nq/64, B*H). 128 thr = 4 waves, 16 query rows each.
// Key tiles of 32 with online softmax; next K/V tile prefetched during
// S-WMMA/softmax/P.V.  Dh = 64 fixed.
// ---------------------------------------------------------------------------
__global__ __launch_bounds__(128)
void attention_kernel(const float* __restrict__ Q, const float* __restrict__ Kx,
                      const float* __restrict__ V, float* __restrict__ O,
                      int Nq, int Mk, int H, int ld, float scale) {
    __shared__ __align__(16) bf16_t sQ [4][16][72];
    __shared__ __align__(16) bf16_t sK [32][72];     // keys x d (== col-major K^T)
    __shared__ __align__(16) bf16_t sVt[64][40];     // d x keys
    __shared__ __align__(16) bf16_t sP [4][16][40];

    const int tid  = threadIdx.x;
    const int lane = tid & 31;
    const int wave = tid >> 5;
    const int bidx = blockIdx.y / H;
    const int h    = blockIdx.y % H;
    const size_t qrow0 = (size_t)bidx * Nq;
    const size_t krow0 = (size_t)bidx * Mk;
    const int col0 = h * 64;
    const int q0   = blockIdx.x * 64;

    // stage Q tile (64x64): batched loads, then stores
    {
        float4 rq[8];
#pragma unroll
        for (int it = 0; it < 8; ++it) {
            const int idx = tid + it * 128;
            const int r   = idx >> 4;
            const int c   = (idx & 15) << 2;
            rq[it] = *(const float4*)(Q + (qrow0 + q0 + r) * ld + col0 + c);
        }
#pragma unroll
        for (int it = 0; it < 8; ++it) {
            const int idx = tid + it * 128;
            const int r   = idx >> 4;
            const int c   = (idx & 15) << 2;
            *(v4bf*)&sQ[r >> 4][r & 15][c] = cvt4(rq[it]);
        }
    }
    __syncthreads();
    const v16bf aq0 = load_frag_a(&sQ[wave][0][0],  72, lane);
    const v16bf aq1 = load_frag_a(&sQ[wave][0][32], 72, lane);

    v8f o[4];
#pragma unroll
    for (int j = 0; j < 4; ++j) o[j] = v8f_zero();
    float mrow[8], lrow[8];
#pragma unroll
    for (int r = 0; r < 8; ++r) { mrow[r] = -1e30f; lrow[r] = 0.0f; }

    const int kq  = tid >> 4;          // key-row quad 0..7
    const int kc4 = (tid & 15) << 2;   // d col 0..60
    float4 rk[4], rv[4];

    auto fetchKV = [&](int kt) {
#pragma unroll
        for (int j = 0; j < 4; ++j) {
            const int r  = 4 * kq + j;
            const int rr = (kt + r < Mk) ? (kt + r) : (Mk - 1);
            rk[j] = *(const float4*)(Kx + (krow0 + rr) * ld + col0 + kc4);
            rv[j] = *(const float4*)(V  + (krow0 + rr) * ld + col0 + kc4);
        }
    };
    auto stageKV = [&](int kt) {
        float4 fv[4];
#pragma unroll
        for (int j = 0; j < 4; ++j) {
            const int r = 4 * kq + j;
            float4 fk = rk[j];
            fv[j] = rv[j];
            if (kt + r >= Mk) {
                fk    = make_float4(0.f, 0.f, 0.f, 0.f);
                fv[j] = make_float4(0.f, 0.f, 0.f, 0.f);
            }
            *(v4bf*)&sK[r][kc4] = cvt4(fk);
        }
#pragma unroll
        for (int j = 0; j < 4; ++j) {   // transposed V, packed along keys
            v4bf hh;
            hh[0] = (bf16_t)f4_at(fv[0], j);
            hh[1] = (bf16_t)f4_at(fv[1], j);
            hh[2] = (bf16_t)f4_at(fv[2], j);
            hh[3] = (bf16_t)f4_at(fv[3], j);
            *(v4bf*)&sVt[kc4 + j][4 * kq] = hh;
        }
    };

    const int MkT = (Mk + 31) & ~31;
    fetchKV(0);
    for (int kt = 0; kt < MkT; kt += 32) {
        stageKV(kt);
        __syncthreads();
        if (kt + 32 < MkT) fetchKV(kt + 32);   // overlap with S/softmax/P.V

        // S(16 x 32 keys) = Q @ K^T
        v8f s0 = v8f_zero(), s1 = v8f_zero();
        {
            v16bf bk;
            bk = load_frag_bt(&sK[0][0],   72, lane); s0 = wmma_bf16(aq0, bk, s0);
            bk = load_frag_bt(&sK[0][32],  72, lane); s0 = wmma_bf16(aq1, bk, s0);
            bk = load_frag_bt(&sK[16][0],  72, lane); s1 = wmma_bf16(aq0, bk, s1);
            bk = load_frag_bt(&sK[16][32], 72, lane); s1 = wmma_bf16(aq1, bk, s1);
        }

        const bool ok0 = (kt +      (lane & 15)) < Mk;
        const bool ok1 = (kt + 16 + (lane & 15)) < Mk;
#pragma unroll
        for (int r = 0; r < 8; ++r) {
            const float e0 = ok0 ? s0[r] * scale : -1e30f;
            const float e1 = ok1 ? s1[r] * scale : -1e30f;
            float t = fmaxf(e0, e1);
            t = fmaxf(t, __shfl_xor(t, 1, 32));
            t = fmaxf(t, __shfl_xor(t, 2, 32));
            t = fmaxf(t, __shfl_xor(t, 4, 32));
            t = fmaxf(t, __shfl_xor(t, 8, 32));
            const float mnew = fmaxf(mrow[r], t);
            const float corr = __expf(mrow[r] - mnew);
            const float p0 = ok0 ? __expf(e0 - mnew) : 0.0f;
            const float p1 = ok1 ? __expf(e1 - mnew) : 0.0f;
            float rsum = p0 + p1;
            rsum += __shfl_xor(rsum, 1, 32);
            rsum += __shfl_xor(rsum, 2, 32);
            rsum += __shfl_xor(rsum, 4, 32);
            rsum += __shfl_xor(rsum, 8, 32);
            lrow[r] = lrow[r] * corr + rsum;
            mrow[r] = mnew;
#pragma unroll
            for (int j = 0; j < 4; ++j) o[j][r] *= corr;
            s0[r] = p0; s1[r] = p1;
        }

        // C-fragment -> A-fragment relayout of P through per-wave LDS region
        const int prow = (lane >> 4) << 3;
        const int pcol = lane & 15;
#pragma unroll
        for (int r = 0; r < 8; ++r) {
            sP[wave][prow + r][pcol]      = (bf16_t)s0[r];
            sP[wave][prow + r][pcol + 16] = (bf16_t)s1[r];
        }
        __syncthreads();

        const v16bf ap = load_frag_a(&sP[wave][0][0], 40, lane);
#pragma unroll
        for (int j = 0; j < 4; ++j) {
            const v16bf bv = load_frag_bt(&sVt[j * 16][0], 40, lane);
            o[j] = wmma_bf16(ap, bv, o[j]);
        }
        __syncthreads();
    }

    const int half = lane >> 4;
    const int ncol = lane & 15;
#pragma unroll
    for (int r = 0; r < 8; ++r) {
        const float inv = 1.0f / lrow[r];
        const int gr = q0 + wave * 16 + half * 8 + r;
#pragma unroll
        for (int j = 0; j < 4; ++j)
            O[(qrow0 + gr) * ld + col0 + j * 16 + ncol] = o[j][r] * inv;
    }
}

// ---------------------------------------------------------------------------
// Fused GEGLU GEMM: block tile 128(M) x 64(N of gated output), 256 thr.
// Dual accumulation of a-half and gate-half of W1; tanh-GELU gate epilogue.
// ---------------------------------------------------------------------------
__global__ __launch_bounds__(256)
void geglu_kernel(const float* __restrict__ A, int lda,
                  const float* __restrict__ W, int ldw,
                  const float* __restrict__ bias,
                  float* __restrict__ G, int ldg,
                  int M, int N, int K, int dff) {
    __shared__ __align__(16) bf16_t sA  [BM][AST];
    __shared__ __align__(16) bf16_t sBat[64][BST];
    __shared__ __align__(16) bf16_t sBgt[64][BST];

    const int tid  = threadIdx.x;
    const int lane = tid & 31;
    const int wave = tid >> 5;
    const int wr   = wave >> 1;
    const int wc   = wave & 1;
    const int m0   = blockIdx.y * BM;
    const int n0   = blockIdx.x * 64;

    // B staging task split: threads 0-127 a-half, 128-255 gate-half.
    const int mat = tid >> 7;
    const int bq  = (tid >> 4) & 7;
    const int bc4 = (tid & 15) << 2;
    const float* Wb = W + (mat ? dff : 0);

    v8f acca[2][2], accg[2][2];
#pragma unroll
    for (int i = 0; i < 2; ++i)
#pragma unroll
        for (int j = 0; j < 2; ++j) { acca[i][j] = v8f_zero(); accg[i][j] = v8f_zero(); }

    float4 ra[4], rb[4];

    auto fetch = [&](int k0) {
#pragma unroll
        for (int it = 0; it < 4; ++it) {
            const int idx = tid + it * 256;
            const int r   = idx >> 3;
            const int c   = (idx & 7) << 2;
            const int rr  = (m0 + r < M) ? (m0 + r) : (M - 1);
            ra[it] = *(const float4*)(A + (size_t)rr * lda + k0 + c);
        }
#pragma unroll
        for (int j = 0; j < 4; ++j)
            rb[j] = *(const float4*)(Wb + (size_t)(k0 + 4 * bq + j) * ldw + n0 + bc4);
    };

    auto stage = [&]() {
#pragma unroll
        for (int it = 0; it < 4; ++it) {
            const int idx = tid + it * 256;
            const int r   = idx >> 3;
            const int c   = (idx & 7) << 2;
            float4 f = ra[it];
            if (m0 + r >= M) f = make_float4(0.f, 0.f, 0.f, 0.f);
            *(v4bf*)&sA[r][c] = cvt4(f);
        }
        bf16_t (*dst)[BST] = mat ? sBgt : sBat;
#pragma unroll
        for (int j = 0; j < 4; ++j) {
            v4bf h;
            h[0] = (bf16_t)f4_at(rb[0], j);
            h[1] = (bf16_t)f4_at(rb[1], j);
            h[2] = (bf16_t)f4_at(rb[2], j);
            h[3] = (bf16_t)f4_at(rb[3], j);
            *(v4bf*)&dst[bc4 + j][4 * bq] = h;
        }
    };

    fetch(0);
    for (int k0 = 0; k0 < K; k0 += BK) {
        stage();
        __syncthreads();
        if (k0 + BK < K) fetch(k0 + BK);
        const v16bf a0 = load_frag_a(&sA[wr * 32][0],      AST, lane);
        const v16bf a1 = load_frag_a(&sA[wr * 32 + 16][0], AST, lane);
#pragma unroll
        for (int j = 0; j < 2; ++j) {
            const v16bf ba = load_frag_bt(&sBat[wc * 32 + j * 16][0], BST, lane);
            const v16bf bg = load_frag_bt(&sBgt[wc * 32 + j * 16][0], BST, lane);
            acca[0][j] = wmma_bf16(a0, ba, acca[0][j]);
            acca[1][j] = wmma_bf16(a1, ba, acca[1][j]);
            accg[0][j] = wmma_bf16(a0, bg, accg[0][j]);
            accg[1][j] = wmma_bf16(a1, bg, accg[1][j]);
        }
        __syncthreads();
    }

    const int half = lane >> 4;
    const int ncol = lane & 15;
#pragma unroll
    for (int i = 0; i < 2; ++i)
#pragma unroll
        for (int j = 0; j < 2; ++j)
#pragma unroll
            for (int r = 0; r < 8; ++r) {
                const int gm = m0 + wr * 32 + i * 16 + half * 8 + r;
                const int gn = n0 + wc * 32 + j * 16 + ncol;
                if (gm < M) {
                    const float ha = acca[i][j][r] + bias[gn];
                    const float hg = accg[i][j][r] + bias[dff + gn];
                    const float t    = 0.7978845608028654f * (hg + 0.044715f * hg * hg * hg);
                    const float gelu = 0.5f * hg * (1.0f + tanhf(t));
                    G[(size_t)gm * ldg + gn] = ha * gelu;
                }
            }
}

// ---------------------------------------------------------------------------
// Host launcher
// ---------------------------------------------------------------------------
static inline dim3 gemm_grid(int M, int N) { return dim3((N + BN - 1) / BN, (M + BM - 1) / BM); }

extern "C" void kernel_launch(void* const* d_in, const int* in_sizes, int n_in,
                              void* d_out, int out_size, void* d_ws, size_t ws_size,
                              hipStream_t stream) {
    (void)in_sizes; (void)n_in; (void)out_size; (void)ws_size;

    const float* x    = (const float*)d_in[0];
    const float* ctx  = (const float*)d_in[1];
    const float* g1   = (const float*)d_in[2];
    const float* be1  = (const float*)d_in[3];
    const float* wq1  = (const float*)d_in[4];
    const float* wk1  = (const float*)d_in[5];
    const float* wv1  = (const float*)d_in[6];
    const float* wo1  = (const float*)d_in[7];
    const float* bo1  = (const float*)d_in[8];
    const float* g2   = (const float*)d_in[9];
    const float* be2  = (const float*)d_in[10];
    const float* wq2  = (const float*)d_in[11];
    const float* wk2  = (const float*)d_in[12];
    const float* wv2  = (const float*)d_in[13];
    const float* wo2  = (const float*)d_in[14];
    const float* bo2  = (const float*)d_in[15];
    const float* g3   = (const float*)d_in[16];
    const float* be3  = (const float*)d_in[17];
    const float* wff1 = (const float*)d_in[18];
    const float* bff1 = (const float*)d_in[19];
    const float* wff2 = (const float*)d_in[20];
    const float* bff2 = (const float*)d_in[21];

    constexpr int Bb = 4, Nq = 1024, D = 1280, CD = 768, Hh = 20, DFF = 5120, Mc = 77;
    const int Rx = Bb * Nq;  // 4096 token rows
    const int Rc = Bb * Mc;  // 308 context rows
    const float scale = 0.125f; // 64^-0.5

    float* ws = (float*)d_ws;
    const size_t S = (size_t)Rx * D;
    float* ln    = ws;
    float* q     = ws + 1 * S;
    float* k     = ws + 2 * S;
    float* v     = ws + 3 * S;
    float* ao    = ws + 4 * S;
    float* x1    = ws + 5 * S;
    float* x2    = ws + 6 * S;
    float* gated = ws + 1 * S; // reuses q..ao (4S == Rx*DFF), dead by then

    const dim3 b256(256), b128(128);

    // ---- self-attention -------------------------------------------------
    layernorm_kernel<<<Rx, 256, 0, stream>>>(x, g1, be1, ln, D);
    gemm_bf16_kernel<<<gemm_grid(Rx, D), b256, 0, stream>>>(ln, D, wq1, D, nullptr, nullptr, q, D, Rx, D, D);
    gemm_bf16_kernel<<<gemm_grid(Rx, D), b256, 0, stream>>>(ln, D, wk1, D, nullptr, nullptr, k, D, Rx, D, D);
    gemm_bf16_kernel<<<gemm_grid(Rx, D), b256, 0, stream>>>(ln, D, wv1, D, nullptr, nullptr, v, D, Rx, D, D);
    attention_kernel<<<dim3(Nq / 64, Bb * Hh), b128, 0, stream>>>(q, k, v, ao, Nq, Nq, Hh, D, scale);
    gemm_bf16_kernel<<<gemm_grid(Rx, D), b256, 0, stream>>>(ao, D, wo1, D, bo1, x, x1, D, Rx, D, D);

    // ---- cross-attention ------------------------------------------------
    layernorm_kernel<<<Rx, 256, 0, stream>>>(x1, g2, be2, ln, D);
    gemm_bf16_kernel<<<gemm_grid(Rx, D), b256, 0, stream>>>(ln, D, wq2, D, nullptr, nullptr, q, D, Rx, D, D);
    gemm_bf16_kernel<<<gemm_grid(Rc, D), b256, 0, stream>>>(ctx, CD, wk2, D, nullptr, nullptr, k, D, Rc, D, CD);
    gemm_bf16_kernel<<<gemm_grid(Rc, D), b256, 0, stream>>>(ctx, CD, wv2, D, nullptr, nullptr, v, D, Rc, D, CD);
    attention_kernel<<<dim3(Nq / 64, Bb * Hh), b128, 0, stream>>>(q, k, v, ao, Nq, Mc, Hh, D, scale);
    gemm_bf16_kernel<<<gemm_grid(Rx, D), b256, 0, stream>>>(ao, D, wo2, D, bo2, x1, x2, D, Rx, D, D);

    // ---- GEGLU FFN ------------------------------------------------------
    layernorm_kernel<<<Rx, 256, 0, stream>>>(x2, g3, be3, ln, D);
    geglu_kernel<<<dim3(DFF / 64, Rx / BM), b256, 0, stream>>>(ln, D, wff1, 2 * DFF, bff1, gated, DFF, Rx, DFF, D, DFF);
    gemm_bf16_kernel<<<gemm_grid(Rx, D), b256, 0, stream>>>(gated, DFF, wff2, D, bff2, x2, (float*)d_out, D, Rx, D, DFF);
}
